// EMAVectorQuantizer_76106820485795
// MI455X (gfx1250) — compile-verified, matrix operand order checked by reading, not run on hardware
//
#include <hip/hip_runtime.h>

typedef __attribute__((ext_vector_type(16))) _Float16 v16h;
typedef __attribute__((ext_vector_type(8)))  float    v8f;

#define NUM_CODES 4096
#define EMBED_DIM 64
#define NVEC      32768          // 8 * 64 * 64
#define ZQ_ELEMS  2097152        // 8 * 64 * 64 * 64
#define ENC_ELEMS 134217728      // NVEC * NUM_CODES

// workspace byte offsets
#define WS_WNORM  0              // float[4096]
#define WS_HIST   16384          // int[4096]
#define WS_LOSS   32768          // float[1] (padded)
#define WS_W16    33024          // uint[4096*32]  (f16-packed codebook)
#define WS_IDX    557312         // int[32768]

// ---------------------------------------------------------------------------
// K0: pack codebook to f16 pairs, compute ||w||^2, zero histogram + loss acc
// ---------------------------------------------------------------------------
__global__ __launch_bounds__(256) void vq_prep(const float* __restrict__ w,
                                               float* __restrict__ wnorm,
                                               unsigned int* __restrict__ w16,
                                               int* __restrict__ hist,
                                               float* __restrict__ lossacc) {
    int code = blockIdx.x * blockDim.x + threadIdx.x;
    if (code >= NUM_CODES) return;
    const float* row = w + code * EMBED_DIM;
    float s = 0.0f;
#pragma unroll
    for (int t = 0; t < 32; ++t) {
        float f0 = row[2 * t];
        float f1 = row[2 * t + 1];
        s += f0 * f0 + f1 * f1;
        union { _Float16 h[2]; unsigned int u; } p;
        p.h[0] = (_Float16)f0;
        p.h[1] = (_Float16)f1;
        w16[code * 32 + t] = p.u;
    }
    wnorm[code] = s;
    hist[code]  = 0;
    if (code == 0) *lossacc = 0.0f;
}

// ---------------------------------------------------------------------------
// K1: zero-fill the 537MB one-hot region with b128 stores.
//     Region base is only 8B-aligned (tuple offset +2 floats), so peel a
//     2-float head and 2-float tail, float4 the aligned interior.
// ---------------------------------------------------------------------------
__global__ __launch_bounds__(256) void vq_zero_enc(float* __restrict__ enc, long n4) {
    long i      = (long)blockIdx.x * blockDim.x + threadIdx.x;
    long stride = (long)gridDim.x * blockDim.x;
    if (blockIdx.x == 0 && threadIdx.x == 0) {
        enc[0] = 0.0f; enc[1] = 0.0f;                       // head
        enc[ENC_ELEMS - 2] = 0.0f; enc[ENC_ELEMS - 1] = 0.0f; // tail
    }
    float4* p = (float4*)(enc + 2);   // 16B-aligned interior
    float4 zz;
    zz.x = 0.0f; zz.y = 0.0f; zz.z = 0.0f; zz.w = 0.0f;
    for (; i < n4; i += stride) p[i] = zz;
}

// ---------------------------------------------------------------------------
// K2: fused GEMM + argmin. One wave per 32-row tile of z (two 16-row A tiles
//     sharing each B tile -> 4 WMMAs per B load, halves L2 traffic).
//     dist = ||w||^2 - 2*(z.w); ||z||^2 constant per row -> dropped.
// ---------------------------------------------------------------------------
__global__ __launch_bounds__(256) void vq_argmin_wmma(const float* __restrict__ z,
                                                      const unsigned int* __restrict__ w16,
                                                      const float* __restrict__ wnorm,
                                                      float* __restrict__ out_idx_f,
                                                      int* __restrict__ ws_idx,
                                                      int* __restrict__ hist) {
    const int lane  = threadIdx.x & 31;
    const int wave  = threadIdx.x >> 5;
    const int tile  = blockIdx.x * 8 + wave;   // 0..1023, 32 rows each
    const int r0    = tile * 32;
    const int lrow  = lane & 15;
    const int khalf = (lane < 16) ? 0 : 8;     // lane-group K split per ISA A-layout

    union V16H { v16h v; unsigned int u[8]; };

    // A matrices (two 16x64 f16 row tiles), ISA lane layout per K=32 block:
    //   elems 0..7  = K kb*32 + khalf + j ; elems 8..15 = K kb*32 + 16 + khalf + j
    V16H a[2][2];
#pragma unroll
    for (int s = 0; s < 2; ++s) {
        const int m = r0 + s * 16 + lrow;
        // z layout [8,64,64,64]: z_flat(m,d) = z[(m>>12)*262144 + d*4096 + (m&4095)]
        const float* zrow = z + (size_t)(m >> 12) * 262144 + (size_t)(m & 4095);
#pragma unroll
        for (int kb = 0; kb < 2; ++kb) {
            const int b0 = kb * 32 + khalf;
            const int b1 = b0 + 16;
#pragma unroll
            for (int j = 0; j < 8; ++j) {
                a[s][kb].v[j]     = (_Float16)zrow[(size_t)(b0 + j) * 4096];
                a[s][kb].v[8 + j] = (_Float16)zrow[(size_t)(b1 + j) * 4096];
            }
        }
    }

    float minv[2][8];
    int   mini[2][8];
#pragma unroll
    for (int s = 0; s < 2; ++s)
#pragma unroll
        for (int j = 0; j < 8; ++j) { minv[s][j] = 3.4e38f; mini[s][j] = 0; }

    const int kh2 = khalf >> 1;   // uint offset of lane's K-half in a 16-uint K=32 block

    for (int c0 = 0; c0 < NUM_CODES; c0 += 16) {
        const int n = c0 + lrow;
        const unsigned int* wrow = w16 + n * 32;   // 64 halves = 32 uints, row-major K
        __builtin_prefetch(wrow + 512, 0, 1);      // next tile's B row -> global_prefetch
        V16H b0v, b1v;
#pragma unroll
        for (int j = 0; j < 4; ++j) {
            b0v.u[j]     = wrow[kh2 + j];           // K = khalf + 2j..+1
            b0v.u[4 + j] = wrow[8 + kh2 + j];       // K = 16 + khalf + 2j
            b1v.u[j]     = wrow[16 + kh2 + j];      // K = 32 + khalf + 2j
            b1v.u[4 + j] = wrow[24 + kh2 + j];      // K = 48 + khalf + 2j
        }
        const float wn = wnorm[n];

        v8f acc0 = {}, acc1 = {};
        acc0 = __builtin_amdgcn_wmma_f32_16x16x32_f16(false, a[0][0].v, false, b0v.v,
                                                      (short)0, acc0, false, false);
        acc1 = __builtin_amdgcn_wmma_f32_16x16x32_f16(false, a[1][0].v, false, b0v.v,
                                                      (short)0, acc1, false, false);
        acc0 = __builtin_amdgcn_wmma_f32_16x16x32_f16(false, a[0][1].v, false, b1v.v,
                                                      (short)0, acc0, false, false);
        acc1 = __builtin_amdgcn_wmma_f32_16x16x32_f16(false, a[1][1].v, false, b1v.v,
                                                      (short)0, acc1, false, false);
#pragma unroll
        for (int j = 0; j < 8; ++j) {
            float d0 = wn - 2.0f * acc0[j];
            float d1 = wn - 2.0f * acc1[j];
            if (d0 < minv[0][j]) { minv[0][j] = d0; mini[0][j] = n; }
            if (d1 < minv[1][j]) { minv[1][j] = d1; mini[1][j] = n; }
        }
    }

    // D layout: lanes 0-15 hold rows +j (VGPR j), lanes 16-31 hold rows +8+j.
    // Branch-free two-phase reduction across each 16-lane half:
    //   1) min-reduce the value; 2) min-reduce (val==vmin ? idx : INT_MAX)
    // (xor masks 1/2/4/8 never cross the lane-16 boundary).
#pragma unroll
    for (int s = 0; s < 2; ++s) {
#pragma unroll
        for (int j = 0; j < 8; ++j) {
            float v = minv[s][j];
#pragma unroll
            for (int off = 1; off < 16; off <<= 1)
                v = fminf(v, __shfl_xor(v, off, 32));
            int cand = (minv[s][j] == v) ? mini[s][j] : 0x7FFFFFFF;
#pragma unroll
            for (int off = 1; off < 16; off <<= 1)
                cand = min(cand, __shfl_xor(cand, off, 32));
            if (lrow == 0) {
                const int row = r0 + s * 16 + j + ((lane >= 16) ? 8 : 0);
                ws_idx[row]    = cand;
                out_idx_f[row] = (float)cand;
                atomicAdd(&hist[cand], 1);
            }
        }
    }
}

// ---------------------------------------------------------------------------
// K3: scatter one-hot, gather z_q (BCHW), accumulate commitment loss
// ---------------------------------------------------------------------------
__global__ __launch_bounds__(256) void vq_scatter(const float* __restrict__ z,
                                                  const float* __restrict__ w,
                                                  const int* __restrict__ ws_idx,
                                                  float* __restrict__ zq,
                                                  float* __restrict__ enc,
                                                  float* __restrict__ lossacc) {
    const int n = blockIdx.x * blockDim.x + threadIdx.x;   // 0..32767
    const int idx = ws_idx[n];
    enc[(size_t)n * NUM_CODES + idx] = 1.0f;
    const float* wrow = w + idx * EMBED_DIM;
    const size_t base = (size_t)(n >> 12) * 262144 + (size_t)(n & 4095);
    float s = 0.0f;
#pragma unroll 8
    for (int c = 0; c < EMBED_DIM; ++c) {
        float q  = wrow[c];
        float zv = z[base + (size_t)c * 4096];
        float df = q - zv;
        s += df * df;
        zq[base + (size_t)c * 4096] = q;   // straight-through forward value == codes
    }
    atomicAdd(lossacc, s);
}

// ---------------------------------------------------------------------------
// K4: perplexity from histogram + final loss scale
// ---------------------------------------------------------------------------
__global__ __launch_bounds__(256) void vq_finalize(const int* __restrict__ hist,
                                                   const float* __restrict__ lossacc,
                                                   float* __restrict__ out_loss,
                                                   float* __restrict__ out_ppl) {
    __shared__ float sdata[256];
    float s = 0.0f;
    for (int c = threadIdx.x; c < NUM_CODES; c += 256) {
        float p = (float)hist[c] * (1.0f / (float)NVEC);
        s += p * logf(p + 1e-10f);
    }
    sdata[threadIdx.x] = s;
    __syncthreads();
    for (int off = 128; off > 0; off >>= 1) {
        if (threadIdx.x < off) sdata[threadIdx.x] += sdata[threadIdx.x + off];
        __syncthreads();
    }
    if (threadIdx.x == 0) {
        *out_ppl  = expf(-sdata[0]);
        *out_loss = 0.25f * (*lossacc) * (1.0f / (float)ZQ_ELEMS);
    }
}

// ---------------------------------------------------------------------------
extern "C" void kernel_launch(void* const* d_in, const int* in_sizes, int n_in,
                              void* d_out, int out_size, void* d_ws, size_t ws_size,
                              hipStream_t stream) {
    const float* z = (const float*)d_in[0];   // [8, 64, 64, 64] fp32
    const float* w = (const float*)d_in[1];   // [4096, 64] fp32

    float* out = (float*)d_out;
    // tuple layout (flat, return order): z_q | loss | perplexity | encodings | indices
    float* out_zq   = out;
    float* out_loss = out + ZQ_ELEMS;
    float* out_ppl  = out + ZQ_ELEMS + 1;
    float* out_enc  = out + ZQ_ELEMS + 2;
    float* out_idx  = out + ZQ_ELEMS + 2 + (size_t)ENC_ELEMS;

    char* ws = (char*)d_ws;
    float*        wnorm   = (float*)(ws + WS_WNORM);
    int*          hist    = (int*)(ws + WS_HIST);
    float*        lossacc = (float*)(ws + WS_LOSS);
    unsigned int* w16     = (unsigned int*)(ws + WS_W16);
    int*          ws_idx  = (int*)(ws + WS_IDX);

    vq_prep<<<NUM_CODES / 256, 256, 0, stream>>>(w, wnorm, w16, hist, lossacc);
    vq_zero_enc<<<4096, 256, 0, stream>>>(out_enc, (long)((ENC_ELEMS - 4) / 4));
    vq_argmin_wmma<<<NVEC / 32 / 8, 256, 0, stream>>>(z, w16, wnorm, out_idx, ws_idx, hist);
    vq_scatter<<<NVEC / 256, 256, 0, stream>>>(z, w, ws_idx, out_zq, out_enc, lossacc);
    vq_finalize<<<1, 256, 0, stream>>>(hist, lossacc, out_loss, out_ppl);
}
